// FatEdgeCentricGNN_32487132627074
// MI455X (gfx1250) — matrix operand order repaired and checked by reading, not compiled, not run on hardware
//
#include <hip/hip_runtime.h>
#include <hip/hip_bf16.h>

typedef _Float16 half_t;
typedef __attribute__((ext_vector_type(16))) _Float16 v16h;
typedef __attribute__((ext_vector_type(8)))  _Float16 v8h;
typedef __attribute__((ext_vector_type(8)))  float    v8f;

#define HIDD   128
#define HEADS  8
#define CDIM   16
#define GGRAPH 128
#define OUTD   6
#define BN_EPS 1e-5f

__device__ __forceinline__ float lrelu_f(float v) { return v > 0.f ? v : 0.2f * v; }

// ------------------------------------------------------------------
// WMMA GEMM, row-strip version: Out[M,Nc] = act(A[M,K] @ Wt[Nc,K]^T + bias)
// A row-major f16, Wt = weight transposed (Nc x K) f16, bias f32.
// One wave computes a 16-row x 128-col strip: 8 accumulator tiles that
// all reuse the same A fragment (8x less HBM traffic on A for the big
// E-row GEMMs that exceed the 192MB L2). All 8 B fragments are preloaded
// per K-step so the load clause drains under the WMMA chain instead of
// stalling each wmma on its own B load. Nc must be a multiple of 128,
// K a multiple of 32.
// ACT: 0 = none, 1 = relu, 2 = leaky relu(0.2)
// ------------------------------------------------------------------
template <int ACT>
__global__ void wmma_gemm(const half_t* __restrict__ A, const half_t* __restrict__ Wt,
                          const float* __restrict__ bias, half_t* __restrict__ Out,
                          int M, int K, int Nc) {
  const int chunks = Nc >> 7;                       // 128-col chunks
  long wave = (long)blockIdx.x * (blockDim.x >> 5) + (threadIdx.x >> 5);
  long tileM = wave / chunks;
  int  chunk = (int)(wave % chunks);
  long rowBase = tileM * 16;
  if (rowBase >= M) return;
  int lane = threadIdx.x & 31;
  int hlf  = lane >> 4;                             // 16-lane half
  int r    = lane & 15;
  long arow = rowBase + r; if (arow >= M) arow = M - 1;   // clamp (stores guarded)
  int colBase = chunk * 128;
  const half_t* Arow = A + arow * (long)K;

  // Per-nt B row base pointers (column = colBase + nt*16 + r)
  const half_t* Brow[8];
#pragma unroll
  for (int nt = 0; nt < 8; ++nt) Brow[nt] = Wt + (long)(colBase + nt * 16 + r) * K + hlf * 16;

  v8f acc[8];
#pragma unroll
  for (int nt = 0; nt < 8; ++nt) acc[nt] = v8f{};

  for (int kk = 0; kk < K; kk += 32) {
    // A 16-bit layout: lane half h holds contiguous K = kk+16h..kk+16h+15
    v8h a0 = *(const v8h*)(Arow + kk + hlf * 16);
    v8h a1 = *(const v8h*)(Arow + kk + hlf * 16 + 8);
    v16h av;
#pragma unroll
    for (int j = 0; j < 8; ++j) { av[j] = a0[j]; av[j + 8] = a1[j]; }
    if (kk + 32 < K) __builtin_prefetch(Arow + kk + 32, 0, 3);

    // Preload ALL B fragments for this K-step (one big load clause),
    // then run the 8-WMMA chain so waits are staggered, not serialized.
    // B 16-bit layout: lane half h holds contiguous K = kk+16h..kk+16h+15.
    v8h b0[8], b1[8];
#pragma unroll
    for (int nt = 0; nt < 8; ++nt) {
      b0[nt] = *(const v8h*)(Brow[nt] + kk);
      b1[nt] = *(const v8h*)(Brow[nt] + kk + 8);
    }
#pragma unroll
    for (int nt = 0; nt < 8; ++nt) {
      v16h bv;
#pragma unroll
      for (int j = 0; j < 8; ++j) { bv[j] = b0[nt][j]; bv[j + 8] = b1[nt][j]; }
      acc[nt] = __builtin_amdgcn_wmma_f32_16x16x32_f16(false, av, false, bv, (short)0, acc[nt],
                                                       false, false);
    }
  }

#pragma unroll
  for (int nt = 0; nt < 8; ++nt) {
    int col = colBase + nt * 16 + r;
    float bval = bias ? bias[col] : 0.f;
#pragma unroll
    for (int i = 0; i < 8; ++i) {
      long row = rowBase + i + hlf * 8;   // C/D layout: VGPR i -> M=i (lanes<16), M=i+8 (lanes>=16)
      if (row < M) {
        float v = acc[nt][i] + bval;
        if (ACT == 1) v = v > 0.f ? v : 0.f;
        else if (ACT == 2) v = lrelu_f(v);
        Out[row * (long)Nc + col] = (half_t)v;
      }
    }
  }
}

// ------------------------------------------------------------------
// Small utility kernels
// ------------------------------------------------------------------
__global__ void prep_weight(const float* __restrict__ W, half_t* __restrict__ Wt,
                            int din, int dout, int kpad) {
  long idx = (long)blockIdx.x * blockDim.x + threadIdx.x;
  if (idx >= (long)dout * kpad) return;
  int n = (int)(idx / kpad), k = (int)(idx % kpad);
  Wt[idx] = (half_t)(k < din ? W[(long)k * dout + n] : 0.f);
}

__global__ void pad_cast_rows(const float* __restrict__ src, half_t* __restrict__ dst,
                              int M, int din, int dpad) {
  long idx = (long)blockIdx.x * blockDim.x + threadIdx.x;
  if (idx >= (long)M * dpad) return;
  int r = (int)(idx / dpad), c = (int)(idx % dpad);
  dst[idx] = (half_t)(c < din ? src[(long)r * din + c] : 0.f);
}

__global__ void fill_f32(float* p, float v, long n) {
  long i = (long)blockIdx.x * blockDim.x + threadIdx.x;
  if (i < n) p[i] = v;
}
__global__ void fill_u32(unsigned* p, unsigned v, long n) {
  long i = (long)blockIdx.x * blockDim.x + threadIdx.x;
  if (i < n) p[i] = v;
}

// ------------------------------------------------------------------
// GINE: message + scatter-add (vectorized b128 feature loads), combine
// ------------------------------------------------------------------
__global__ void gine_msg(const half_t* __restrict__ h, const half_t* __restrict__ e,
                         const int* __restrict__ ei, int E, float* __restrict__ agg) {
  long idx = (long)blockIdx.x * blockDim.x + threadIdx.x;
  if (idx >= (long)E * 16) return;
  int eid = (int)(idx >> 4);
  int c0  = ((int)idx & 15) * 8;
  int s = ei[eid], d = ei[E + eid];
  v8h hv = *(const v8h*)(h + (long)s * HIDD + c0);
  v8h ev = *(const v8h*)(e + (long)eid * HIDD + c0);
#pragma unroll
  for (int j = 0; j < 8; ++j) {
    float m = (float)hv[j] + (float)ev[j];
    if (m > 0.f) atomicAdd(&agg[(long)d * HIDD + c0 + j], m);   // relu: skip non-positive
  }
}

__global__ void gine_combine(const half_t* __restrict__ h, const float* __restrict__ agg,
                             const float* __restrict__ epsp, half_t* __restrict__ z, long total) {
  long i = (long)blockIdx.x * blockDim.x + threadIdx.x;
  if (i >= total) return;
  float e1 = 1.f + epsp[0];
  z[i] = (half_t)(e1 * (float)h[i] + agg[i]);
}

__global__ void add_f16(const half_t* a, const half_t* b, half_t* c, long total) {
  long i = (long)blockIdx.x * blockDim.x + threadIdx.x;
  if (i >= total) return;
  c[i] = (half_t)((float)a[i] + (float)b[i]);
}

// att(f32) + skip(f16) -> lrelu -> f16
__global__ void attn_combine(const float* att, const half_t* s, half_t* outp, long total) {
  long i = (long)blockIdx.x * blockDim.x + threadIdx.x;
  if (i >= total) return;
  outp[i] = (half_t)lrelu_f(att[i] + (float)s[i]);
}

// ------------------------------------------------------------------
// BatchNorm (training mode, gamma=1 beta=0): column stats + apply
// ------------------------------------------------------------------
__global__ void bn_stats(const half_t* __restrict__ X, int M, float* mu, float* rstd) {
  __shared__ float ss[256], ss2[256];
  int col = blockIdx.x;
  float s = 0.f, s2 = 0.f;
  for (int i = threadIdx.x; i < M; i += 256) {
    float v = (float)X[(long)i * HIDD + col];
    s += v; s2 += v * v;
  }
  ss[threadIdx.x] = s; ss2[threadIdx.x] = s2;
  __syncthreads();
  for (int off = 128; off > 0; off >>= 1) {
    if (threadIdx.x < off) { ss[threadIdx.x] += ss[threadIdx.x + off]; ss2[threadIdx.x] += ss2[threadIdx.x + off]; }
    __syncthreads();
  }
  if (threadIdx.x == 0) {
    float m = ss[0] / M;
    float var = ss2[0] / M - m * m;
    mu[col] = m;
    rstd[col] = rsqrtf(var + BN_EPS);
  }
}

__global__ void bn_apply(const half_t* __restrict__ X, half_t* __restrict__ Y,
                         const float* __restrict__ mu, const float* __restrict__ rstd, long total) {
  long i = (long)blockIdx.x * blockDim.x + threadIdx.x;
  if (i >= total) return;
  int col = (int)(i & (HIDD - 1));
  Y[i] = (half_t)(((float)X[i] - mu[col]) * rstd[col]);
}

// ------------------------------------------------------------------
// TransformerConv attention (segment softmax over dst via atomics)
// ------------------------------------------------------------------
__device__ __forceinline__ unsigned enc_key(float f) {
  int b = __float_as_int(f);
  return (unsigned)(b ^ ((b >> 31) | 0x80000000));
}
__device__ __forceinline__ float dec_key(unsigned k) {
  int b = (k & 0x80000000u) ? (int)(k ^ 0x80000000u) : ~(int)k;
  return __int_as_float(b);
}

__global__ void attn_alpha(const half_t* __restrict__ q, const half_t* __restrict__ k,
                           const half_t* __restrict__ ee, const int* __restrict__ ei, int E,
                           float* __restrict__ alpha, unsigned* __restrict__ amax) {
  long idx = (long)blockIdx.x * blockDim.x + threadIdx.x;
  if (idx >= (long)E * HEADS) return;
  int eid = (int)(idx >> 3), hd = (int)idx & 7;
  int s = ei[eid], d = ei[E + eid];
  v8h q0 = *(const v8h*)(q + (long)d * HIDD + hd * CDIM);
  v8h q1 = *(const v8h*)(q + (long)d * HIDD + hd * CDIM + 8);
  v8h k0 = *(const v8h*)(k + (long)s * HIDD + hd * CDIM);
  v8h k1 = *(const v8h*)(k + (long)s * HIDD + hd * CDIM + 8);
  v8h e0 = *(const v8h*)(ee + (long)eid * HIDD + hd * CDIM);
  v8h e1 = *(const v8h*)(ee + (long)eid * HIDD + hd * CDIM + 8);
  float sum = 0.f;
#pragma unroll
  for (int c = 0; c < 8; ++c) {
    sum += (float)q0[c] * ((float)k0[c] + (float)e0[c]);
    sum += (float)q1[c] * ((float)k1[c] + (float)e1[c]);
  }
  float a = sum * 0.25f;   // 1/sqrt(16)
  alpha[idx] = a;
  atomicMax(&amax[(long)d * HEADS + hd], enc_key(a));
}

__global__ void attn_exp(const int* __restrict__ ei, int E, float* __restrict__ alpha,
                         const unsigned* __restrict__ amax, float* __restrict__ den) {
  long idx = (long)blockIdx.x * blockDim.x + threadIdx.x;
  if (idx >= (long)E * HEADS) return;
  int eid = (int)(idx >> 3), hd = (int)idx & 7;
  int d = ei[E + eid];
  float ex = expf(alpha[idx] - dec_key(amax[(long)d * HEADS + hd]));
  alpha[idx] = ex;
  atomicAdd(&den[(long)d * HEADS + hd], ex);
}

__global__ void attn_agg(const half_t* __restrict__ v, const half_t* __restrict__ ee,
                         const int* __restrict__ ei, int E, const float* __restrict__ alpha,
                         const float* __restrict__ den, float* __restrict__ att) {
  long idx = (long)blockIdx.x * blockDim.x + threadIdx.x;
  if (idx >= (long)E * HEADS) return;
  int eid = (int)(idx >> 3), hd = (int)idx & 7;
  int s = ei[eid], d = ei[E + eid];
  float a = alpha[idx] / (den[(long)d * HEADS + hd] + 1e-16f);
  v8h v0 = *(const v8h*)(v + (long)s * HIDD + hd * CDIM);
  v8h v1 = *(const v8h*)(v + (long)s * HIDD + hd * CDIM + 8);
  v8h e0 = *(const v8h*)(ee + (long)eid * HIDD + hd * CDIM);
  v8h e1 = *(const v8h*)(ee + (long)eid * HIDD + hd * CDIM + 8);
#pragma unroll
  for (int c = 0; c < 8; ++c) {
    atomicAdd(&att[(long)d * HIDD + hd * CDIM + c],     ((float)v0[c] + (float)e0[c]) * a);
    atomicAdd(&att[(long)d * HIDD + hd * CDIM + 8 + c], ((float)v1[c] + (float)e1[c]) * a);
  }
}

// ------------------------------------------------------------------
// Pooling + FC head
// ------------------------------------------------------------------
__global__ void pool_sum(const half_t* __restrict__ h, const int* __restrict__ batch, int N,
                         float* __restrict__ hg, float* __restrict__ cnt) {
  long idx = (long)blockIdx.x * blockDim.x + threadIdx.x;
  if (idx >= (long)N * HIDD) return;
  int n = (int)(idx >> 7), c = (int)idx & 127;
  int b = batch[n];
  atomicAdd(&hg[(long)b * HIDD + c], (float)h[idx]);
  if (c == 0) atomicAdd(&cnt[b], 1.f);
}

__global__ void pool_div(float* hg, const float* cnt) {
  int idx = blockIdx.x * blockDim.x + threadIdx.x;
  if (idx >= GGRAPH * HIDD) return;
  hg[idx] /= fmaxf(cnt[idx >> 7], 1.f);
}

__global__ void fc_head(const float* __restrict__ hg,
                        const float* W0, const float* B0, const float* W1, const float* B1,
                        const float* W2, const float* B2, const float* W3, const float* B3,
                        float* __restrict__ out) {
  int g = blockIdx.x, t = threadIdx.x;
  __shared__ float h0[128], h1[256], h2[12], h3[12];
  if (t < 128) h0[t] = hg[(long)g * HIDD + t];
  __syncthreads();
  if (t < 256) { float s = B0[t]; for (int i = 0; i < 128; ++i) s += h0[i] * W0[(long)i * 256 + t]; h1[t] = s > 0.f ? s : 0.f; }
  __syncthreads();
  if (t < 12) { float s = B1[t]; for (int i = 0; i < 256; ++i) s += h1[i] * W1[(long)i * 12 + t]; h2[t] = s > 0.f ? s : 0.f; }
  __syncthreads();
  if (t < 12) { float s = B2[t]; for (int i = 0; i < 12; ++i) s += h2[i] * W2[i * 12 + t]; h3[t] = s > 0.f ? s : 0.f; }
  __syncthreads();
  if (t < 6) { float s = B3[t]; for (int i = 0; i < 12; ++i) s += h3[i] * W3[i * 6 + t]; out[g * OUTD + t] = s; }
}

// ------------------------------------------------------------------
// Host orchestration
// ------------------------------------------------------------------
extern "C" void kernel_launch(void* const* d_in, const int* in_sizes, int n_in,
                              void* d_out, int out_size, void* d_ws, size_t ws_size,
                              hipStream_t stream) {
  (void)n_in; (void)out_size; (void)ws_size;
  const int N = in_sizes[0] / 32;        // node dim 32
  const int E = in_sizes[2] / 16;        // edge dim 16

  const float* x     = (const float*)d_in[0];
  const int*   ei    = (const int*)d_in[1];     // [2,E]: src then dst
  const float* ea    = (const float*)d_in[2];
  const int*   batch = (const int*)d_in[3];

  // ---- param index bookkeeping (setup_inputs dict insertion order, recursive) ----
  int idx = 4;
  int i_ne0W = idx++, i_ne0b = idx++, i_ne1W = idx++, i_ne1b = idx++;
  int i_ee0W = idx++, i_ee0b = idx++, i_ee1W = idx++, i_ee1b = idx++, i_ee2W = idx++, i_ee2b = idx++;
  int i_gl1W[11], i_gl1b[11], i_gl2W[11], i_gl2b[11], i_geps[11];
  for (int g = 0; g < 11; ++g) { i_gl1W[g]=idx++; i_gl1b[g]=idx++; i_gl2W[g]=idx++; i_gl2b[g]=idx++; i_geps[g]=idx++; }
  int i_tqW[3], i_tqb[3], i_tkW[3], i_tkb[3], i_tvW[3], i_tvb[3], i_tWe[3], i_tsW[3], i_tsb[3];
  for (int t = 0; t < 3; ++t) { i_tqW[t]=idx++; i_tqb[t]=idx++; i_tkW[t]=idx++; i_tkb[t]=idx++;
                                i_tvW[t]=idx++; i_tvb[t]=idx++; i_tWe[t]=idx++; i_tsW[t]=idx++; i_tsb[t]=idx++; }
  int i_skW[4], i_skb[4];
  for (int s = 0; s < 4; ++s) { i_skW[s]=idx++; i_skb[s]=idx++; }
  int i_fcW[4], i_fcb[4];
  for (int f = 0; f < 4; ++f) { i_fcW[f]=idx++; i_fcb[f]=idx++; }

  // ---- workspace carve ----
  char* ws = (char*)d_ws;
  size_t cur = 0;
  auto alloc = [&](size_t bytes) -> char* {
    char* p = ws + cur;
    cur = (cur + bytes + 255) & ~(size_t)255;
    return p;
  };
  half_t* t1   = (half_t*)alloc((size_t)E * 256 * 2);   // edge-enc hidden 1; later reused as ee16
  half_t* t2   = (half_t*)alloc((size_t)E * 256 * 2);   // edge-enc hidden 2; early: ea16; later: alpha
  half_t* e16  = (half_t*)alloc((size_t)E * HIDD * 2);  // encoded edge features (persistent)
  half_t* x16  = (half_t*)alloc((size_t)N * 32 * 2);
  half_t* hA   = (half_t*)alloc((size_t)N * HIDD * 2);
  half_t* hB   = (half_t*)alloc((size_t)N * HIDD * 2);
  half_t* hIn  = (half_t*)alloc((size_t)N * HIDD * 2);  // h_init for skips
  half_t* zb   = (half_t*)alloc((size_t)N * HIDD * 2);
  half_t* tb   = (half_t*)alloc((size_t)N * HIDD * 2);
  half_t* gb   = (half_t*)alloc((size_t)N * HIDD * 2);
  half_t* sb   = (half_t*)alloc((size_t)N * HIDD * 2);
  half_t* q16  = (half_t*)alloc((size_t)N * HIDD * 2);
  half_t* k16  = (half_t*)alloc((size_t)N * HIDD * 2);
  half_t* v16  = (half_t*)alloc((size_t)N * HIDD * 2);
  float*  agg  = (float*)alloc((size_t)N * HIDD * 4);
  float*  att  = (float*)alloc((size_t)N * HIDD * 4);
  float*    den   = (float*)alloc((size_t)N * HEADS * 4);
  unsigned* amaxk = (unsigned*)alloc((size_t)N * HEADS * 4);
  float*  mu   = (float*)alloc(HIDD * 4);
  float*  rstd = (float*)alloc(HIDD * 4);
  float*  hg   = (float*)alloc((size_t)GGRAPH * HIDD * 4);
  float*  cntb = (float*)alloc(GGRAPH * 4);
  half_t* ea16 = (half_t*)t2;                 // alias (dead before t2 written)
  half_t* ee16 = (half_t*)t1;                 // alias (t1 dead after encoder)
  float*  alph = (float*)t2;                  // alias (t2 dead after encoder)

  // ---- helpers ----
  auto gemm = [&](const half_t* A, const half_t* Wt, const float* bias, half_t* Out,
                  long M, int K, int Nc, int act) {
    long waves = ((M + 15) / 16) * (Nc / 128);
    int blocks = (int)((waves + 7) / 8);
    if (act == 0)      wmma_gemm<0><<<blocks, 256, 0, stream>>>(A, Wt, bias, Out, (int)M, K, Nc);
    else if (act == 1) wmma_gemm<1><<<blocks, 256, 0, stream>>>(A, Wt, bias, Out, (int)M, K, Nc);
    else               wmma_gemm<2><<<blocks, 256, 0, stream>>>(A, Wt, bias, Out, (int)M, K, Nc);
  };
  auto prep = [&](int srcIdx, int din, int dout, int kpad) -> half_t* {
    half_t* dst = (half_t*)alloc((size_t)dout * kpad * 2);
    long tot = (long)dout * kpad;
    prep_weight<<<(int)((tot + 255) / 256), 256, 0, stream>>>((const float*)d_in[srcIdx], dst, din, dout, kpad);
    return dst;
  };
  auto bf = [&](int i) { return (const float*)d_in[i]; };
  auto blocks1d = [](long n) { return (int)((n + 255) / 256); };
  const long NH = (long)N * HIDD;

  // ---- weight prep (f16, transposed, K padded to mult of 32) ----
  half_t* w_ne0 = prep(i_ne0W, 32, 128, 32);
  half_t* w_ne1 = prep(i_ne1W, 128, 128, 128);
  half_t* w_ee0 = prep(i_ee0W, 16, 256, 32);
  half_t* w_ee1 = prep(i_ee1W, 256, 256, 256);
  half_t* w_ee2 = prep(i_ee2W, 256, 128, 256);
  half_t *w_gl1[11], *w_gl2[11];
  for (int g = 0; g < 11; ++g) { w_gl1[g] = prep(i_gl1W[g], 128, 128, 128); w_gl2[g] = prep(i_gl2W[g], 128, 128, 128); }
  half_t *w_tq[3], *w_tk[3], *w_tv[3], *w_te[3], *w_ts[3];
  for (int t = 0; t < 3; ++t) {
    w_tq[t] = prep(i_tqW[t], 128, 128, 128); w_tk[t] = prep(i_tkW[t], 128, 128, 128);
    w_tv[t] = prep(i_tvW[t], 128, 128, 128); w_te[t] = prep(i_tWe[t], 128, 128, 128);
    w_ts[t] = prep(i_tsW[t], 128, 128, 128);
  }
  half_t* w_sk[4];
  for (int s = 0; s < 4; ++s) w_sk[s] = prep(i_skW[s], 128, 128, 128);

  // ---- encoders ----
  pad_cast_rows<<<blocks1d((long)E * 32), 256, 0, stream>>>(ea, ea16, E, 16, 32);
  pad_cast_rows<<<blocks1d((long)N * 32), 256, 0, stream>>>(x, x16, N, 32, 32);
  gemm(ea16, w_ee0, bf(i_ee0b), t1, E, 32, 256, 1);
  gemm(t1, w_ee1, bf(i_ee1b), t2, E, 256, 256, 1);
  gemm(t2, w_ee2, bf(i_ee2b), e16, E, 256, 128, 0);
  gemm(x16, w_ne0, bf(i_ne0b), zb, N, 32, 128, 1);
  gemm(zb, w_ne1, bf(i_ne1b), hA, N, 128, 128, 0);

  half_t* hcur = hA;
  half_t* hnxt = hB;
  auto do_bn = [&](const half_t* src) {
    bn_stats<<<HIDD, 256, 0, stream>>>(src, N, mu, rstd);
    bn_apply<<<blocks1d(NH), 256, 0, stream>>>(src, hnxt, mu, rstd, NH);
    half_t* t0 = hcur; hcur = hnxt; hnxt = t0;
    if (hnxt == src) hnxt = (hcur == hA) ? hB : hA;   // keep hnxt a real rotating buffer
  };
  auto run_gine = [&](int g, bool bn) {
    fill_f32<<<blocks1d(NH), 256, 0, stream>>>(agg, 0.f, NH);
    gine_msg<<<blocks1d((long)E * 16), 256, 0, stream>>>(hcur, e16, ei, E, agg);
    gine_combine<<<blocks1d(NH), 256, 0, stream>>>(hcur, agg, bf(i_geps[g]), zb, NH);
    gemm(zb, w_gl1[g], bf(i_gl1b[g]), tb, N, 128, 128, 1);
    gemm(tb, w_gl2[g], bf(i_gl2b[g]), gb, N, 128, 128, 2);   // lrelu fused
    if (bn) do_bn(gb);
    else hcur = gb;                                          // c4c: lrelu only
  };
  auto run_skip = [&](int s) {
    gemm(hIn, w_sk[s], bf(i_skb[s]), sb, N, 128, 128, 0);
    add_f16<<<blocks1d(NH), 256, 0, stream>>>(hcur, sb, tb, NH);
    do_bn(tb);
  };
  auto run_tc = [&](int t) {
    gemm(hcur, w_tq[t], bf(i_tqb[t]), q16, N, 128, 128, 0);
    gemm(hcur, w_tk[t], bf(i_tkb[t]), k16, N, 128, 128, 0);
    gemm(hcur, w_tv[t], bf(i_tvb[t]), v16, N, 128, 128, 0);
    gemm(e16, w_te[t], nullptr, ee16, E, 128, 128, 0);       // big E x 128 x 128 WMMA GEMM
    fill_u32<<<blocks1d((long)N * HEADS), 256, 0, stream>>>(amaxk, 0u, (long)N * HEADS);
    fill_f32<<<blocks1d((long)N * HEADS), 256, 0, stream>>>(den, 0.f, (long)N * HEADS);
    fill_f32<<<blocks1d(NH), 256, 0, stream>>>(att, 0.f, NH);
    attn_alpha<<<blocks1d((long)E * HEADS), 256, 0, stream>>>(q16, k16, ee16, ei, E, alph, amaxk);
    attn_exp<<<blocks1d((long)E * HEADS), 256, 0, stream>>>(ei, E, alph, amaxk, den);
    attn_agg<<<blocks1d((long)E * HEADS), 256, 0, stream>>>(v16, ee16, ei, E, alph, den, att);
    gemm(hcur, w_ts[t], bf(i_tsb[t]), sb, N, 128, 128, 0);
    attn_combine<<<blocks1d(NH), 256, 0, stream>>>(att, sb, tb, NH);
    do_bn(tb);
  };
  auto snap_hinit = [&]() { hipMemcpyAsync(hIn, hcur, NH * 2, hipMemcpyDeviceToDevice, stream); };

  // ---- block 2 ----
  snap_hinit();
  run_gine(0, true); run_gine(1, true); run_gine(2, true);
  run_skip(0);
  run_tc(0);
  // ---- block 3 ----
  snap_hinit();
  run_gine(3, true); run_gine(4, true); run_gine(5, true);
  run_skip(1);
  run_tc(1);
  // ---- block 4 ----
  snap_hinit();
  run_gine(6, true); run_gine(7, true); run_gine(8, false);  // c4c: no BN
  run_skip(2);
  // ---- block 5 ----
  snap_hinit();
  run_gine(9, true); run_gine(10, true);
  run_skip(3);
  run_tc(2);

  // ---- pooling + FC head ----
  fill_f32<<<blocks1d((long)GGRAPH * HIDD), 256, 0, stream>>>(hg, 0.f, (long)GGRAPH * HIDD);
  fill_f32<<<1, 256, 0, stream>>>(cntb, 0.f, GGRAPH);
  pool_sum<<<blocks1d(NH), 256, 0, stream>>>(hcur, batch, N, hg, cntb);
  pool_div<<<(GGRAPH * HIDD + 255) / 256, 256, 0, stream>>>(hg, cntb);
  fc_head<<<GGRAPH, 256, 0, stream>>>(hg,
      bf(i_fcW[0]), bf(i_fcb[0]), bf(i_fcW[1]), bf(i_fcb[1]),
      bf(i_fcW[2]), bf(i_fcb[2]), bf(i_fcW[3]), bf(i_fcb[3]),
      (float*)d_out);
}
